// PrototypeLearner_53377853555142
// MI455X (gfx1250) — matrix operand order repaired and checked by reading, not compiled, not run on hardware
//
#include <hip/hip_runtime.h>
#include <hip/hip_fp16.h>

typedef __attribute__((ext_vector_type(16))) _Float16 v16h;
typedef __attribute__((ext_vector_type(8)))  _Float16 v8h;
typedef __attribute__((ext_vector_type(4)))  _Float16 v4h;
typedef __attribute__((ext_vector_type(8)))  float    v8f;

#define NUM_CLASSES 200
#define NUM_PROTO   10
#define EMBED_DIM   512
#define QUEUE_SIZE  8192
#define CAP         256      // max tokens per class (mean 41, ~33 sigma headroom)
#define ITERS       5
#define EPS_SINK    0.01f
#define MOMENTUM    0.02f

// ---------------- Kernel 1: L2-normalize tokens, emit f16 copy + inv norms ----
// grid = QUEUE_SIZE blocks, 128 threads (1 float4 per thread)
__global__ void __launch_bounds__(128) k_norm(const float* __restrict__ tokens,
                                              float* __restrict__ inv_norm,
                                              _Float16* __restrict__ tokh) {
  const int t = blockIdx.x;
  const int tid = threadIdx.x;
  const float4 v = ((const float4*)(tokens + (size_t)t * EMBED_DIM))[tid];
  float ss = v.x*v.x + v.y*v.y + v.z*v.z + v.w*v.w;
  #pragma unroll
  for (int off = 16; off > 0; off >>= 1) ss += __shfl_xor(ss, off, 32);
  __shared__ float sred[4];
  if ((tid & 31) == 0) sred[tid >> 5] = ss;
  __syncthreads();
  const float inv = rsqrtf(sred[0] + sred[1] + sred[2] + sred[3]);
  if (tid == 0) inv_norm[t] = inv;
  v4h h = { (_Float16)(v.x * inv), (_Float16)(v.y * inv),
            (_Float16)(v.z * inv), (_Float16)(v.w * inv) };
  *(v4h*)(tokh + (size_t)t * EMBED_DIM + tid * 4) = h;   // one b64 store
}

// ---------------- Kernel 2: deterministic per-class token index lists --------
// grid = NUM_CLASSES blocks, 256 threads. Stable order => reproducible FP sums.
__global__ void __launch_bounds__(256) k_index(const int* __restrict__ labels,
                                               int* __restrict__ counts,
                                               int* __restrict__ order) {
  const int c = blockIdx.x;
  const int tid = threadIdx.x;
  __shared__ int sl[256];
  __shared__ int scnt;
  if (tid == 0) scnt = 0;
  for (int base = 0; base < QUEUE_SIZE; base += 256) {
    sl[tid] = labels[base + tid];
    __syncthreads();
    if (tid == 0) {
      int cnt = scnt;
      #pragma unroll 4
      for (int k = 0; k < 256; k++)
        if (sl[k] == c && cnt < CAP) order[c * CAP + cnt++] = base + k;
      scnt = cnt;
    }
    __syncthreads();
  }
  if (tid == 0) counts[c] = scnt;
}

// ---------------- Kernel 3: per-class WMMA GEMM + masked Sinkhorn + update ---
// grid = NUM_CLASSES blocks, 256 threads (8 wave32).
__global__ void __launch_bounds__(256) k_sinkhorn(
    const float*    __restrict__ tokens,
    const float*    __restrict__ prototypes,
    const float*    __restrict__ inv_norm,
    const _Float16* __restrict__ tokh,
    const int*      __restrict__ counts,
    const int*      __restrict__ order,
    float*          __restrict__ out) {
  const int c    = blockIdx.x;
  const int tid  = threadIdx.x;
  const int lane = tid & 31;
  const int wave = tid >> 5;

  __shared__ _Float16 sA[16][EMBED_DIM];        // padded f16 prototypes (16 KB)
  __shared__ float sS[16][CAP];                 // similarity, later pi_ (16 KB)
  __shared__ float sBlend[NUM_PROTO][EMBED_DIM];// 20 KB
  __shared__ float sU[16];
  __shared__ float sV[CAP];
  __shared__ int   sG[CAP];
  __shared__ float sRN[NUM_PROTO];

  int n = counts[c]; if (n > CAP) n = CAP;

  for (int j = tid; j < CAP; j += 256) {
    sG[j] = (j < n) ? order[c * CAP + j] : 0;
    sV[j] = 0.f;
  }
  if (tid < 16) sU[tid] = 0.f;
  // stage A: rows 0..9 = f16(protos of class c), rows 10..15 = 0
  for (int idx = tid; idx < 16 * EMBED_DIM; idx += 256) {
    const int row = idx >> 9, k = idx & 511;
    sA[row][k] = (row < NUM_PROTO)
        ? (_Float16)prototypes[(size_t)(c * NUM_PROTO + row) * EMBED_DIM + k]
        : (_Float16)0.f;
  }
  __syncthreads();

  // === S[16][n] = protos_c @ tok_c^T via v_wmma_f32_16x16x32_f16 ===
  // Padded rows/cols produce dead sS entries that are never read, so no
  // fragment masking is needed (indices are clamped to stay in-bounds).
  const int ntile = (n + 15) >> 4;
  const int r  = lane & 15;          // A row / C column-in-tile
  const int hi = lane >> 4;          // lane half selector
  const _Float16* apA = &sA[r][hi * 16];

  for (int t = wave; t < ntile; t += 8) {
    v8f acc = {};
    const int j = (t << 4) + r;
    const _Float16* bp = tokh + (size_t)sG[(j < n) ? j : 0] * EMBED_DIM + hi * 16;
    #pragma unroll 4
    for (int kk = 0; kk < EMBED_DIM; kk += 32) {
      // A (16-bit A 16x32): lane holds K in [kk+hi*16,+16) of row r
      // (same k order as B)  -> two ds_load_b128 from the staged tile.
      const v8h a0 = *(const v8h*)(apA + kk);
      const v8h a1 = *(const v8h*)(apA + kk + 8);
      v16h a;
      #pragma unroll
      for (int e = 0; e < 8; e++) { a[e] = a0[e]; a[8 + e] = a1[e]; }
      // B (dense 32x16): lane holds K in [kk+hi*16,+16) of its column.
      const v16h b = *(const v16h*)(bp + kk);
      acc = __builtin_amdgcn_wmma_f32_16x16x32_f16(
          false, a, false, b, (short)0, acc, false, false);
    }
    // C layout: VGPR vr, lane -> row vr + 8*hi, col = tile*16 + (lane&15)
    const int col = (t << 4) + r;
    #pragma unroll
    for (int vr = 0; vr < 8; vr++) sS[vr + 8 * hi][col] = acc[vr];
  }
  __syncthreads();

  // === masked Sinkhorn, decomposed per class (mask is block-diagonal) ===
  const float inv_eps = 1.0f / EPS_SINK;
  const float log_a = logf(1.0f / (float)NUM_PROTO + 1e-8f);
  const float log_b = logf(1.0f / (float)(n < 1 ? 1 : n) + 1e-8f);

  for (int it = 0; it < ITERS; it++) {
    // u update: row-wise logsumexp over this class's n columns
    for (int i = wave; i < NUM_PROTO; i += 8) {
      const float u = sU[i];
      float m = -__builtin_inff();
      for (int j = lane; j < n; j += 32)
        m = fmaxf(m, (sS[i][j] + u + sV[j]) * inv_eps);
      #pragma unroll
      for (int off = 16; off > 0; off >>= 1) m = fmaxf(m, __shfl_xor(m, off, 32));
      float s = 0.f;
      for (int j = lane; j < n; j += 32)
        s += __expf((sS[i][j] + u + sV[j]) * inv_eps - m);
      #pragma unroll
      for (int off = 16; off > 0; off >>= 1) s += __shfl_xor(s, off, 32);
      if (lane == 0) {
        float u_ = log_a - (m + __logf(s));
        if (!isfinite(u_)) u_ = 0.f;            // nan_to_num(posinf=0)
        sU[i] = EPS_SINK * u_ + u;
      }
    }
    __syncthreads();
    // v update: column-wise logsumexp over the 10 same-class rows
    for (int j = tid; j < n; j += 256) {
      const float v = sV[j];
      float m = -__builtin_inff();
      #pragma unroll
      for (int i = 0; i < NUM_PROTO; i++)
        m = fmaxf(m, (sS[i][j] + sU[i] + v) * inv_eps);
      float s = 0.f;
      #pragma unroll
      for (int i = 0; i < NUM_PROTO; i++)
        s += __expf((sS[i][j] + sU[i] + v) * inv_eps - m);
      float v_ = log_b - (m + __logf(s));
      if (!isfinite(v_)) v_ = 0.f;
      sV[j] = EPS_SINK * v_ + v;
    }
    __syncthreads();
  }

  // pi = exp(kernel); column-normalize (off-class rows contribute 0) in place
  for (int j = tid; j < n; j += 256) {
    const float v = sV[j];
    float e[NUM_PROTO], sum = 0.f;
    #pragma unroll
    for (int i = 0; i < NUM_PROTO; i++) {
      e[i] = __expf((sS[i][j] + sU[i] + v) * inv_eps);
      sum += e[i];
    }
    const float inv = (sum > 0.f) ? 1.0f / sum : 0.f;
    #pragma unroll
    for (int i = 0; i < NUM_PROTO; i++) sS[i][j] = e[i] * inv;
  }
  __syncthreads();

  // new_protos = pi_ @ tok_c (f32), momentum blend
  for (int idx = tid; idx < NUM_PROTO * EMBED_DIM; idx += 256) {
    const int i = idx >> 9, d = idx & 511;
    float acc = 0.f;
    for (int j = 0; j < n; j++) {
      const int g = sG[j];
      acc += sS[i][j] * (tokens[(size_t)g * EMBED_DIM + d] * inv_norm[g]);
    }
    const float p = prototypes[(size_t)(c * NUM_PROTO + i) * EMBED_DIM + d];
    sBlend[i][d] = (1.0f - MOMENTUM) * p + MOMENTUM * acc;
  }
  __syncthreads();

  // final per-row L2 renormalization
  for (int i = wave; i < NUM_PROTO; i += 8) {
    float ss = 0.f;
    for (int d = lane; d < EMBED_DIM; d += 32) { float x = sBlend[i][d]; ss += x * x; }
    #pragma unroll
    for (int off = 16; off > 0; off >>= 1) ss += __shfl_xor(ss, off, 32);
    if (lane == 0) sRN[i] = rsqrtf(ss);
  }
  __syncthreads();
  for (int idx = tid; idx < NUM_PROTO * EMBED_DIM; idx += 256) {
    const int i = idx >> 9, d = idx & 511;
    out[(size_t)(c * NUM_PROTO + i) * EMBED_DIM + d] = sBlend[i][d] * sRN[i];
  }
}

// ---------------------------------------------------------------------------
extern "C" void kernel_launch(void* const* d_in, const int* in_sizes, int n_in,
                              void* d_out, int out_size, void* d_ws, size_t ws_size,
                              hipStream_t stream) {
  const float* tokens     = (const float*)d_in[0];
  const int*   labels     = (const int*)d_in[1];
  const float* prototypes = (const float*)d_in[2];
  float*       out        = (float*)d_out;

  char* ws = (char*)d_ws;
  float*    inv_norm = (float*)(ws);             // 8192*4      = 32 KB
  int*      counts   = (int*)(ws + 32768);       // 200*4 (pad 4 KB)
  int*      order    = (int*)(ws + 36864);       // 200*256*4   = 200 KB
  _Float16* tokh     = (_Float16*)(ws + 241664); // 8192*512*2  = 8 MB

  k_norm<<<QUEUE_SIZE, 128, 0, stream>>>(tokens, inv_norm, tokh);
  k_index<<<NUM_CLASSES, 256, 0, stream>>>(labels, counts, order);
  k_sinkhorn<<<NUM_CLASSES, 256, 0, stream>>>(tokens, prototypes, inv_norm, tokh,
                                              counts, order, out);
}
